// VQVAE_38190849196517
// MI455X (gfx1250) — matrix operand order repaired and checked by reading, not compiled, not run on hardware
//
#include <hip/hip_runtime.h>
#include <hip/hip_bf16.h>
#include <stdint.h>

typedef __attribute__((ext_vector_type(16))) _Float16 v16h;
typedef __attribute__((ext_vector_type(8)))  float    v8f;

#if defined(__gfx1250__) && \
    __has_builtin(__builtin_amdgcn_global_load_async_to_lds_b32) && \
    __has_builtin(__builtin_amdgcn_s_wait_asynccnt)
#define HAVE_ASYNC_LDS 1
typedef __attribute__((address_space(1))) int gas_int;
typedef __attribute__((address_space(3))) int las_int;
#else
#define HAVE_ASYNC_LDS 0
#endif

namespace {

constexpr int kB   = 32;
constexpr int kT   = 1024;
constexpr int kCI  = 261;   // J*D = 87*3
constexpr int kH   = 512;
constexpr int kCB  = 768;
constexpr int kCD  = 64;
constexpr int kOR  = 256;
constexpr int kT2  = 512;
constexpr int kT4  = 256;

constexpr int kChunk = 32;   // K (ci) chunk per WMMA
constexpr int kTileT = 32;   // t-tile per block (2 x 16 per wave)
constexpr int kRowF  = 36;   // padded LDS row stride (floats): 144B, 16B-aligned

__host__ __device__ inline int ceilDivI(int a, int b) { return (a + b - 1) / b; }
inline int pad64(int x) { return ceilDivI(x, 64) * 64; }
inline int pad32(int x) { return ceilDivI(x, 32) * 32; }

// ---------------------------------------------------------------- transposes
__global__ __launch_bounds__(256)
void k_tx_btc_to_bct(const float* __restrict__ src, float* __restrict__ dst,
                     int Bn, int Tn, int Cn) {
  size_t i = (size_t)blockIdx.x * blockDim.x + threadIdx.x;
  size_t total = (size_t)Bn * Tn * Cn;
  if (i >= total) return;
  int t = (int)(i % Tn);
  size_t r = i / Tn;
  int c = (int)(r % Cn);
  int b = (int)(r / Cn);
  dst[i] = src[((size_t)b * Tn + t) * Cn + c];
}

__global__ __launch_bounds__(256)
void k_tx_bct_to_btc(const float* __restrict__ src, float* __restrict__ dst,
                     int Bn, int Cn, int Tn) {
  size_t i = (size_t)blockIdx.x * blockDim.x + threadIdx.x;
  size_t total = (size_t)Bn * Tn * Cn;
  if (i >= total) return;
  int c = (int)(i % Cn);
  size_t r = i / Cn;
  int t = (int)(r % Tn);
  int b = (int)(r / Tn);
  dst[i] = src[((size_t)b * Cn + c) * Tn + t];
}

// ------------------------------------------------- weight prep: f32 -> f16 wT
// dst layout: [k][co (pad64)][ci (pad32)], zero padded.
// srcIOK==0: src is [co][ci][k]   (Conv1d)
// srcIOK==1: src is [ci][co][k]   (ConvTranspose1d)
__global__ __launch_bounds__(256)
void k_prep_w(const float* __restrict__ src, _Float16* __restrict__ dst,
              int Co, int Ci, int K, int CoPad, int CiPad, int srcIOK) {
  size_t i = (size_t)blockIdx.x * blockDim.x + threadIdx.x;
  size_t total = (size_t)K * CoPad * CiPad;
  if (i >= total) return;
  int ci = (int)(i % CiPad);
  size_t r = i / CiPad;
  int co = (int)(r % CoPad);
  int k  = (int)(r / CoPad);
  float v = 0.0f;
  if (co < Co && ci < Ci)
    v = srcIOK ? src[((size_t)ci * Co + co) * K + k]
               : src[((size_t)co * Ci + ci) * K + k];
  dst[i] = (_Float16)v;
}

__global__ __launch_bounds__(256)
void k_l2norm_rows(const float* __restrict__ src, float* __restrict__ dst,
                   int N, int D) {
  int n = blockIdx.x * blockDim.x + threadIdx.x;
  if (n >= N) return;
  float s = 0.0f;
  for (int d = 0; d < D; ++d) { float v = src[(size_t)n * D + d]; s += v * v; }
  float r = rsqrtf(s + 1e-12f);
  for (int d = 0; d < D; ++d) dst[(size_t)n * D + d] = src[(size_t)n * D + d] * r;
}

// ------------------------------------------------------- generic WMMA conv1d
// 4-wave block computes a 256(co) x 32(t) macro-tile. The 32ci x 32t f32
// activation tile is staged into double-buffered LDS with async global->LDS
// copies, prefetching chunk i+1 while WMMAs consume chunk i (one barrier per
// chunk). Each wave holds two B fragments (t columns 0-15 / 16-31) and reuses
// every A (weight) fragment across both -> 8 WMMAs per chunk per wave.
__global__ __launch_bounds__(128)
void k_conv_wmma(const float* __restrict__ in, int Tin, int CinA,
                 float* __restrict__ out, int Tout, int Cout,
                 const _Float16* __restrict__ wT, int CiPad, int CoPad,
                 const float* __restrict__ bias,
                 int ntap, int4 tapW, int4 tapO,
                 int inStride, int outStride, int outOff, int Nt,
                 int reluIn, int reluOut, const float* __restrict__ resid) {
  __shared__ __align__(16) float tile[2][kTileT * kRowF];

  const int tid  = threadIdx.x;
  const int wid  = tid >> 5;
  const int lane = tid & 31;
  const int hg   = lane >> 4;
  const int lm   = lane & 15;
  const int t0   = blockIdx.x * kTileT;
  const int coW  = blockIdx.y * 256 + wid * 64;   // this wave's co base
  const int b    = blockIdx.z;

  const float* inB = in + (size_t)b * CinA * Tin;
  const int tw[4] = {tapW.x, tapW.y, tapW.z, tapW.w};
  const int to[4] = {tapO.x, tapO.y, tapO.z, tapO.w};

  const int nIter = ntap * (CiPad / kChunk);

  v8f acc0[4] = {};
  v8f acc1[4] = {};
  const int tI0 = t0 + lm;        // column set 0
  const int tI1 = t0 + 16 + lm;   // column set 1

  // staging mapping: thread covers (t = tid&31, ci = (tid>>5) + 4*r), r=0..7
  const int sT  = tid & 31;
  const int sC  = tid >> 5;       // 0..3
  const int sTt = t0 + sT;

  int tapS = 0, ci0S = 0;         // staging (prefetch) coordinates
  int tapC = 0, ci0C = 0;         // compute coordinates

  auto stage = [&](int bufIdx) {
    const int  tinS = sTt * inStride + to[tapS];
    const bool sOk  = (sTt < Nt) && (tinS >= 0) && (tinS < Tin);
    float* bufp = &tile[bufIdx][0];
#pragma unroll
    for (int r = 0; r < 8; ++r) {
      const int ciLoc = sC + 4 * r;           // 0..31
      const int ci    = ci0S + ciLoc;
      float* lp = &bufp[sT * kRowF + ciLoc];
      if (sOk && ci < CinA) {
        const float* gp = &inB[(size_t)ci * Tin + tinS];
#if HAVE_ASYNC_LDS
        __builtin_amdgcn_global_load_async_to_lds_b32(
            (gas_int*)gp, (las_int*)lp, 0, 0);
#else
        *lp = *gp;
#endif
      } else {
        *lp = 0.0f;
      }
    }
    ci0S += kChunk;
    if (ci0S >= CiPad) { ci0S = 0; ++tapS; }
  };

  stage(0);
#if HAVE_ASYNC_LDS
  __builtin_amdgcn_s_wait_asynccnt(0);
#endif
  __syncthreads();

  for (int it = 0; it < nIter; ++it) {
    const int cur = it & 1;
    if (it + 1 < nIter) stage(cur ^ 1);   // async prefetch next chunk

    // ---- B fragments from LDS (element j <-> K = hg*16 + j) ------------
    const float* bufp = &tile[cur][0];
    union { uint4 q[4]; float f[16]; } Bt0, Bt1;
    {
      const uint4* q0 = (const uint4*)(bufp + lm * kRowF + hg * 16);
      const uint4* q1 = (const uint4*)(bufp + (lm + 16) * kRowF + hg * 16);
      Bt0.q[0] = q0[0]; Bt0.q[1] = q0[1]; Bt0.q[2] = q0[2]; Bt0.q[3] = q0[3];
      Bt1.q[0] = q1[0]; Bt1.q[1] = q1[1]; Bt1.q[2] = q1[2]; Bt1.q[3] = q1[3];
    }
    union { v16h v; _Float16 h[16]; } Bf0, Bf1;
#pragma unroll
    for (int j = 0; j < 16; ++j) {
      float x0 = Bt0.f[j], x1 = Bt1.f[j];
      if (reluIn) { x0 = fmaxf(x0, 0.0f); x1 = fmaxf(x1, 0.0f); }
      Bf0.h[j] = (_Float16)x0;
      Bf1.h[j] = (_Float16)x1;
    }

    // ---- A fragments; reuse each across both B fragments ---------------
    const _Float16* wTap = wT + (size_t)tw[tapC] * CoPad * CiPad;
#pragma unroll
    for (int s = 0; s < 4; ++s) {
      const int co16 = coW + s * 16;          // wave-uniform guard
      if (co16 < CoPad) {
        const _Float16* wRow = wTap + (size_t)(co16 + lm) * CiPad + ci0C;
        union { v16h v; uint4 q[2]; } Af;
        Af.q[0] = *(const uint4*)(wRow + hg * 16);       // K = hg*16 + 0..7
        Af.q[1] = *(const uint4*)(wRow + hg * 16 + 8);   // K = hg*16 + 8..15
        acc0[s] = __builtin_amdgcn_wmma_f32_16x16x32_f16(
            false, Af.v, false, Bf0.v, (short)0, acc0[s], false, false);
        acc1[s] = __builtin_amdgcn_wmma_f32_16x16x32_f16(
            false, Af.v, false, Bf1.v, (short)0, acc1[s], false, false);
      }
    }
    ci0C += kChunk;
    if (ci0C >= CiPad) { ci0C = 0; ++tapC; }

#if HAVE_ASYNC_LDS
    if (it + 1 < nIter) __builtin_amdgcn_s_wait_asynccnt(0);
#endif
    __syncthreads();
  }

  // ---- epilogue: C/D f32 16x16: VGPR r -> M=r+8*hg, lane%16 -> N -------
#pragma unroll
  for (int s = 0; s < 4; ++s) {
#pragma unroll
    for (int r = 0; r < 8; ++r) {
      const int c = coW + s * 16 + r + 8 * hg;
      if (c < Cout) {
        const size_t rowBase = ((size_t)b * Cout + c) * Tout;
        if (tI0 < Nt) {
          float v = acc0[s][r];
          if (bias) v += bias[c];
          const size_t oi = rowBase + (size_t)tI0 * outStride + outOff;
          if (resid) v += resid[oi];
          if (reluOut) v = fmaxf(v, 0.0f);
          out[oi] = v;
        }
        if (tI1 < Nt) {
          float v = acc1[s][r];
          if (bias) v += bias[c];
          const size_t oi = rowBase + (size_t)tI1 * outStride + outOff;
          if (resid) v += resid[oi];
          if (reluOut) v = fmaxf(v, 0.0f);
          out[oi] = v;
        }
      }
    }
  }
}

// ------------------------------------------------------------------- VQ bits
__global__ __launch_bounds__(256)
void k_argmax(const float* __restrict__ sim, int* __restrict__ idx,
              int Bn, int N, int Tn) {
  int i = blockIdx.x * blockDim.x + threadIdx.x;
  if (i >= Bn * Tn) return;
  int t = i % Tn, b = i / Tn;
  const float* p = sim + (size_t)b * N * Tn + t;
  float best = p[0]; int bi = 0;
  for (int n = 1; n < N; ++n) {
    float v = p[(size_t)n * Tn];
    if (v > best) { best = v; bi = n; }
  }
  idx[i] = bi;
}

__global__ __launch_bounds__(256)
void k_gather_q(const float* __restrict__ en, const int* __restrict__ idx,
                float* __restrict__ q, int Bn, int Tn, int D) {
  size_t i = (size_t)blockIdx.x * blockDim.x + threadIdx.x;
  size_t total = (size_t)Bn * D * Tn;
  if (i >= total) return;
  int t = (int)(i % Tn);
  size_t r = i / Tn;
  int d = (int)(r % D);
  int b = (int)(r / D);
  q[i] = en[(size_t)idx[b * Tn + t] * D + d];
}

__global__ __launch_bounds__(256)
void k_commit(const float* __restrict__ xi, const float* __restrict__ en,
              const int* __restrict__ idx, float* __restrict__ part,
              int Bn, int D, int Tn) {
  int i = blockIdx.x * blockDim.x + threadIdx.x;
  if (i >= Bn * Tn) return;
  int t = i % Tn, b = i / Tn;
  const float* xp = xi + (size_t)b * D * Tn + t;
  const float* ep = en + (size_t)idx[i] * D;
  float s = 0.0f;
  for (int d = 0; d < D; ++d) {
    float df = ep[d] - xp[(size_t)d * Tn];
    s += df * df;
  }
  part[i] = s;
}

__global__ __launch_bounds__(256)
void k_orth(const float* __restrict__ en, float* __restrict__ part, int D) {
  __shared__ float sh[256];
  int i = blockIdx.x, j = threadIdx.x;
  float s = 0.0f;
  for (int d = 0; d < D; ++d) s += en[(size_t)i * D + d] * en[(size_t)j * D + d];
  sh[j] = s * s;
  __syncthreads();
  for (int k = 128; k > 0; k >>= 1) {
    if (j < k) sh[j] += sh[j + k];
    __syncthreads();
  }
  if (j == 0) part[i] = sh[0];
}

__global__ void k_final(const float* __restrict__ cpart, int nc,
                        const float* __restrict__ opart, int no,
                        float* __restrict__ out) {
  float cs = 0.0f;
  for (int i = 0; i < nc; ++i) cs += cpart[i];
  float os = 0.0f;
  for (int i = 0; i < no; ++i) os += opart[i];
  float commit = (cs / (float)((size_t)nc * kCD)) * 0.9f;
  float orth   = os / ((float)no * (float)no) - 1.0f / (float)no;
  out[0] = commit + orth;
}

} // namespace

extern "C" void kernel_launch(void* const* d_in, const int* in_sizes, int n_in,
                              void* d_out, int out_size, void* d_ws, size_t ws_size,
                              hipStream_t stream) {
  (void)in_sizes; (void)out_size; (void)ws_size;
  const float* IN[66] = {};
  for (int i = 0; i < 66 && i < n_in; ++i) IN[i] = (const float*)d_in[i];

  char* ws = (char*)d_ws;
  size_t off = 0;
  auto alloc = [&](size_t bytes) -> char* {
    char* p = ws + off;
    off = (off + bytes + 255) & ~(size_t)255;
    return p;
  };
  auto allocW = [&](int Co, int Ci, int K) -> _Float16* {
    return (_Float16*)alloc((size_t)K * pad64(Co) * pad32(Ci) * sizeof(_Float16));
  };
  auto prep = [&](const float* src, _Float16* dst, int Co, int Ci, int K, int iok) {
    size_t total = (size_t)K * pad64(Co) * pad32(Ci);
    k_prep_w<<<dim3((unsigned)((total + 255) / 256)), 256, 0, stream>>>(
        src, dst, Co, Ci, K, pad64(Co), pad32(Ci), iok);
  };
  auto conv = [&](const float* in, int Tin, int CinA, float* out, int Tout,
                  int Cout, const _Float16* w, int Ci, int Co,
                  const float* bias, int ntap, int4 tw, int4 to, int inStride,
                  int outStride, int outOff, int Nt, int reluIn, int reluOut,
                  const float* resid) {
    dim3 g((unsigned)ceilDivI(Nt, kTileT), (unsigned)ceilDivI(Cout, 256), kB);
    k_conv_wmma<<<g, 128, 0, stream>>>(in, Tin, CinA, out, Tout, Cout, w,
                                       pad32(Ci), pad64(Co), bias, ntap, tw, to,
                                       inStride, outStride, outOff, Nt, reluIn,
                                       reluOut, resid);
  };
  auto resblk = [&](float* x, float* tmp, int Tc, const _Float16* w1,
                    const float* b1, const _Float16* w2, const float* b2,
                    int dil) {
    conv(x, Tc, kH, tmp, Tc, kH, w1, kH, kH, b1, 3, make_int4(0, 1, 2, 0),
         make_int4(-dil, 0, dil, 0), 1, 1, 0, Tc, 1, 0, nullptr);
    conv(tmp, Tc, kH, x, Tc, kH, w2, kH, kH, b2, 1, make_int4(0, 0, 0, 0),
         make_int4(0, 0, 0, 0), 1, 1, 0, Tc, 1, 0, x);
  };

  // -------- input index map (setup_inputs insertion-order flatten) --------
  const float* x     = IN[0];
  const float* wInW  = IN[1];
  const float* cOutW = IN[59]; const float* cOutB = IN[60];
  const float* piW   = IN[61]; const float* piB   = IN[62];
  const float* poW   = IN[63]; const float* poB   = IN[64];
  const float* embed = IN[65];

  // -------- weight prep (f32 -> padded f16 [tap][co][ci]) --------
  _Float16* WIN = allocW(kH, kCI, 3);  prep(wInW, WIN, kH, kCI, 3, 0);
  _Float16* WDN[2]; _Float16* WE1[2][3]; _Float16* WE2[2][3];
  for (int bi = 0; bi < 2; ++bi) {
    WDN[bi] = allocW(kH, kH, 4);
    prep(IN[3 + bi * 14], WDN[bi], kH, kH, 4, 0);
    for (int ri = 0; ri < 3; ++ri) {
      int base = 5 + bi * 14 + ri * 4;
      WE1[bi][ri] = allocW(kH, kH, 3); prep(IN[base + 0], WE1[bi][ri], kH, kH, 3, 0);
      WE2[bi][ri] = allocW(kH, kH, 1); prep(IN[base + 2], WE2[bi][ri], kH, kH, 1, 0);
    }
  }
  _Float16* WUP[2]; _Float16* WD1[2][3]; _Float16* WD2[2][3];
  for (int bi = 0; bi < 2; ++bi) {
    for (int ri = 0; ri < 3; ++ri) {
      int base = 31 + bi * 14 + ri * 4;
      WD1[bi][ri] = allocW(kH, kH, 3); prep(IN[base + 0], WD1[bi][ri], kH, kH, 3, 0);
      WD2[bi][ri] = allocW(kH, kH, 1); prep(IN[base + 2], WD2[bi][ri], kH, kH, 1, 0);
    }
    WUP[bi] = allocW(kH, kH, 4);
    prep(IN[43 + bi * 14], WUP[bi], kH, kH, 4, 1);  // ConvT weight is [ci][co][k]
  }
  _Float16* WOUT = allocW(kCI, kH, 3); prep(cOutW, WOUT, kCI, kH, 3, 0);
  _Float16* WPI  = allocW(kCD, kH, 1); prep(piW, WPI, kCD, kH, 1, 0);
  _Float16* WPO  = allocW(kH, kCD, 1); prep(poW, WPO, kH, kCD, 1, 0);

  float* enF = (float*)alloc((size_t)kCB * kCD * sizeof(float));
  k_l2norm_rows<<<dim3((unsigned)ceilDivI(kCB, 256)), 256, 0, stream>>>(embed, enF, kCB, kCD);
  _Float16* WEN = allocW(kCB, kCD, 1); prep(enF, WEN, kCB, kCD, 1, 0);

  // -------- activation / scratch buffers --------
  const size_t actBytes = (size_t)kB * kH * kT * sizeof(float);
  float* bufA = (float*)alloc(actBytes);
  float* bufB = (float*)alloc(actBytes);
  float* bufC = (float*)alloc(actBytes);
  float* xiB  = (float*)alloc((size_t)kB * kCD * kT4 * sizeof(float));
  float* qB   = (float*)alloc((size_t)kB * kCD * kT4 * sizeof(float));
  float* cPart = (float*)alloc((size_t)kB * kT4 * sizeof(float));
  float* oPart = (float*)alloc((size_t)kOR * sizeof(float));

  const int4 z4 = make_int4(0, 0, 0, 0);
  const int dilsE[3] = {9, 3, 1};
  const int dilsD[3] = {1, 3, 9};

  // -------- encoder --------
  {
    size_t total = (size_t)kB * kCI * kT;
    k_tx_btc_to_bct<<<dim3((unsigned)((total + 255) / 256)), 256, 0, stream>>>(
        x, bufB, kB, kT, kCI);
  }
  conv(bufB, kT, kCI, bufA, kT, kH, WIN, kCI, kH, IN[2], 3,
       make_int4(0, 1, 2, 0), make_int4(-1, 0, 1, 0), 1, 1, 0, kT, 0, 1, nullptr);
  conv(bufA, kT, kH, bufB, kT2, kH, WDN[0], kH, kH, IN[4], 4,
       make_int4(0, 1, 2, 3), make_int4(-1, 0, 1, 2), 2, 1, 0, kT2, 0, 0, nullptr);
  for (int ri = 0; ri < 3; ++ri) {
    int base = 5 + 0 * 14 + ri * 4;
    resblk(bufB, bufC, kT2, WE1[0][ri], IN[base + 1], WE2[0][ri], IN[base + 3], dilsE[ri]);
  }
  conv(bufB, kT2, kH, bufA, kT4, kH, WDN[1], kH, kH, IN[18], 4,
       make_int4(0, 1, 2, 3), make_int4(-1, 0, 1, 2), 2, 1, 0, kT4, 0, 0, nullptr);
  for (int ri = 0; ri < 3; ++ri) {
    int base = 5 + 1 * 14 + ri * 4;
    resblk(bufA, bufC, kT4, WE1[1][ri], IN[base + 1], WE2[1][ri], IN[base + 3], dilsE[ri]);
  }

  // -------- VQ --------
  conv(bufA, kT4, kH, xiB, kT4, kCD, WPI, kH, kCD, piB, 1, z4, z4,
       1, 1, 0, kT4, 0, 0, nullptr);
  conv(xiB, kT4, kCD, bufC, kT4, kCB, WEN, kCD, kCB, nullptr, 1, z4, z4,
       1, 1, 0, kT4, 0, 0, nullptr);
  const size_t reconN = (size_t)kB * kT * kCI;
  int* idxOut = (int*)((float*)d_out + reconN);
  k_argmax<<<dim3((unsigned)ceilDivI(kB * kT4, 256)), 256, 0, stream>>>(
      bufC, idxOut, kB, kCB, kT4);
  {
    size_t total = (size_t)kB * kCD * kT4;
    k_gather_q<<<dim3((unsigned)((total + 255) / 256)), 256, 0, stream>>>(
        enF, idxOut, qB, kB, kT4, kCD);
  }
  k_commit<<<dim3((unsigned)ceilDivI(kB * kT4, 256)), 256, 0, stream>>>(
      xiB, enF, idxOut, cPart, kB, kCD, kT4);
  k_orth<<<dim3(kOR), 256, 0, stream>>>(enF, oPart, kCD);
  conv(qB, kT4, kCD, bufA, kT4, kH, WPO, kCD, kH, poB, 1, z4, z4,
       1, 1, 0, kT4, 0, 0, nullptr);

  // -------- decoder --------
  for (int ri = 0; ri < 3; ++ri) {
    int base = 31 + 0 * 14 + ri * 4;
    resblk(bufA, bufC, kT4, WD1[0][ri], IN[base + 1], WD2[0][ri], IN[base + 3], dilsD[ri]);
  }
  conv(bufA, kT4, kH, bufB, kT2, kH, WUP[0], kH, kH, IN[44], 2,
       make_int4(1, 3, 0, 0), make_int4(0, -1, 0, 0), 1, 2, 0, kT4, 0, 0, nullptr);
  conv(bufA, kT4, kH, bufB, kT2, kH, WUP[0], kH, kH, IN[44], 2,
       make_int4(0, 2, 0, 0), make_int4(1, 0, 0, 0), 1, 2, 1, kT4, 0, 0, nullptr);
  for (int ri = 0; ri < 3; ++ri) {
    int base = 31 + 1 * 14 + ri * 4;
    resblk(bufB, bufC, kT2, WD1[1][ri], IN[base + 1], WD2[1][ri], IN[base + 3], dilsD[ri]);
  }
  conv(bufB, kT2, kH, bufA, kT, kH, WUP[1], kH, kH, IN[58], 2,
       make_int4(1, 3, 0, 0), make_int4(0, -1, 0, 0), 1, 2, 0, kT2, 0, 0, nullptr);
  conv(bufB, kT2, kH, bufA, kT, kH, WUP[1], kH, kH, IN[58], 2,
       make_int4(0, 2, 0, 0), make_int4(1, 0, 0, 0), 1, 2, 1, kT2, 0, 0, nullptr);
  conv(bufA, kT, kH, bufB, kT, kCI, WOUT, kH, kCI, cOutB, 3,
       make_int4(0, 1, 2, 0), make_int4(-1, 0, 1, 0), 1, 1, 0, kT, 0, 0, nullptr);
  k_tx_bct_to_btc<<<dim3((unsigned)((reconN + 255) / 256)), 256, 0, stream>>>(
      bufB, (float*)d_out, kB, kCI, kT);

  // -------- loss scalar --------
  k_final<<<1, 1, 0, stream>>>(cPart, kB * kT4, oPart, kOR,
                               (float*)d_out + reconN + (size_t)kB * kT4);
}
